// GraphBlock_57930518888703
// MI455X (gfx1250) — matrix operand order repaired and checked by reading, not compiled, not run on hardware
//
#include <hip/hip_runtime.h>
#include <stdint.h>

#define Bc  64
#define Nn  512
#define Fd  64
#define Hd  128
#define FOd 64
#define Kn  16
#define BIGV 1000000000.0f

// padded LDS row strides (in bf16 elements) to spread banks:
// 136 ush = 272B -> dword step 68 %64 = 4  -> 16 distinct banks
// 200 ush = 400B -> dword step 100 %64 = 36 -> 16 distinct banks
#define PAD128 136
#define PAD192 200

typedef __attribute__((ext_vector_type(16))) __bf16 v16bf;
typedef __attribute__((ext_vector_type(8)))  float  v8f;

union FragU { v16bf v; uint4 u[2]; };

__device__ __forceinline__ unsigned short f2bf(float x) {
  unsigned int u = __float_as_uint(x);
  u += 0x7FFFu + ((u >> 16) & 1u);          // round-to-nearest-even
  return (unsigned short)(u >> 16);
}

// A-matrix 16x32 bf16 fragment (ISA 7.12.2): lane l<16: M=l, elems0-7 K=k0..k0+7,
// elems8-15 K=k0+16..k0+23; lanes 16-31: same M, K ranges +8. Row-major LDS, lda elems.
__device__ __forceinline__ v16bf load_a_frag(const unsigned short* sA, int lda,
                                             int row0, int k0, int lane) {
  int h = lane >> 4, r = lane & 15;
  const unsigned short* p = sA + (row0 + r) * lda + k0 + h * 8;
  FragU f;
  f.u[0] = *(const uint4*)(p);
  f.u[1] = *(const uint4*)(p + 16);
  return f.v;
}

// B-matrix 32x16 bf16 fragment: lanes 0-15 hold K=k0..k0+7 and k0+16..k0+23 of col N=lane,
// lanes 16-31 hold K ranges +8 of col N=lane-16. Weights stored transposed
// [n][k] (padded stride) so each 8 K-value group per lane is contiguous.
__device__ __forceinline__ v16bf load_b_frag(const unsigned short* sWt, int ldk,
                                             int col0, int k0, int lane) {
  int h = lane >> 4, n = lane & 15;
  const unsigned short* p = sWt + (col0 + n) * ldk + k0 + h * 8;
  FragU f;
  f.u[0] = *(const uint4*)(p);
  f.u[1] = *(const uint4*)(p + 16);
  return f.v;
}

// ---------------- prep: x = nodes*mask, sq = ||x||^2, bf16 copy ----------------
__global__ void prep_x_kernel(const float* __restrict__ nodes, const int* __restrict__ mask,
                              float* __restrict__ x, unsigned short* __restrict__ xbf,
                              float* __restrict__ sq) {
  int wave = threadIdx.x >> 5, lane = threadIdx.x & 31;
  int row = blockIdx.x * 8 + wave;             // node row in [0, B*N)
  float mval = (float)mask[row];
  float v0 = nodes[row * Fd + lane] * mval;
  float v1 = nodes[row * Fd + lane + 32] * mval;
  x[row * Fd + lane] = v0;
  x[row * Fd + lane + 32] = v1;
  xbf[row * Fd + lane] = f2bf(v0);
  xbf[row * Fd + lane + 32] = f2bf(v1);
  float s = v0 * v0 + v1 * v1;
  #pragma unroll
  for (int o = 16; o >= 1; o >>= 1) s += __shfl_xor(s, o, 32);
  if (lane == 0) sq[row] = s;
}

// ---------------- prep: transpose weights to [n][k] bf16 ----------------
__global__ void wprep_kernel(const float* __restrict__ W, unsigned short* __restrict__ Wt,
                             int Kdim, int Ndim) {
  int i = blockIdx.x * blockDim.x + threadIdx.x;
  if (i >= Kdim * Ndim) return;
  int n = i / Kdim, k = i % Kdim;
  Wt[i] = f2bf(W[k * Ndim + n]);
}

// ---------------- top-K: one wave per row, 16 candidates/lane ----------------
__global__ void topk_kernel(const float* __restrict__ x, const float* __restrict__ sq,
                            const int* __restrict__ mask,
                            int* __restrict__ idx, float* __restrict__ emask,
                            float* __restrict__ cnt) {
  __shared__ float sxn[8][Fd];
  int wave = threadIdx.x >> 5, lane = threadIdx.x & 31;
  int row = blockIdx.x * 8 + wave;
  int b = row / Nn;
  sxn[wave][lane]      = x[row * Fd + lane];
  sxn[wave][lane + 32] = x[row * Fd + lane + 32];
  __syncthreads();
  float sqn = sq[row];
  float mn  = (float)mask[row];

  float d[16];
  #pragma unroll
  for (int c = 0; c < 16; ++c) {
    int m = c * 32 + lane;
    const float4* xm = (const float4*)(x + (size_t)(b * Nn + m) * Fd);
    float dot = 0.f;
    #pragma unroll
    for (int f = 0; f < 16; ++f) {
      float4 a = xm[f];
      dot += a.x * sxn[wave][f * 4 + 0] + a.y * sxn[wave][f * 4 + 1]
           + a.z * sxn[wave][f * 4 + 2] + a.w * sxn[wave][f * 4 + 3];
    }
    float dm = fabsf(sqn + sq[b * Nn + m] - 2.f * dot);
    d[c] = dm + BIGV * (1.f - (float)mask[b * Nn + m]);
  }

  float emv = 0.f;
  #pragma unroll 1
  for (int r = 0; r < Kn + 1; ++r) {            // K+1 smallest, drop first (self)
    float best = 3.0e38f; int bs = 0;
    #pragma unroll
    for (int c = 0; c < 16; ++c) if (d[c] < best) { best = d[c]; bs = c; }
    unsigned int khi = __float_as_uint(best);   // dist >= 0 -> bits monotonic
    unsigned int klo = (unsigned int)(bs * 32 + lane);
    #pragma unroll
    for (int o = 16; o >= 1; o >>= 1) {
      unsigned int ohi = __shfl_xor(khi, o, 32);
      unsigned int olo = __shfl_xor(klo, o, 32);
      if (ohi < khi || (ohi == khi && olo < klo)) { khi = ohi; klo = olo; }
    }
    int mwin = (int)klo, owner = mwin & 31, slot = mwin >> 5;
    #pragma unroll
    for (int c = 0; c < 16; ++c) if (lane == owner && c == slot) d[c] = 3.0e38f;
    if (r > 0 && lane == r - 1) {
      idx[row * Kn + (r - 1)] = mwin;
      emv = mn * (float)mask[b * Nn + mwin];
      emask[row * Kn + (r - 1)] = emv;
    }
  }
  float csum = emv;
  #pragma unroll
  for (int o = 16; o >= 1; o >>= 1) csum += __shfl_xor(csum, o, 32);
  if (lane == 0) cnt[row] = fmaxf(csum, 1.f);
}

// ---------------- edge MLP: 8 nodes/block (128 edge rows), two 128^3 WMMA GEMMs ---------
__global__ void edge_mlp_kernel(const unsigned short* __restrict__ xbf,
                                const int* __restrict__ idx,
                                const unsigned short* __restrict__ We1t,
                                const unsigned short* __restrict__ We2t,
                                const float* __restrict__ be1, const float* __restrict__ be2,
                                const float* __restrict__ emask, const float* __restrict__ cnt,
                                float* __restrict__ pooled) {
  __shared__ unsigned short sW0[Hd * PAD128];   // We1^T, padded
  __shared__ unsigned short sW1[Hd * PAD128];   // We2^T, padded
  __shared__ unsigned short sA[128 * PAD128];   // edge matrix / h1, padded
  int tid = threadIdx.x, wave = tid >> 5, lane = tid & 31;
  int h = lane >> 4;
  int nodeBase = blockIdx.x * 8;
  int node = nodeBase + wave;

  __builtin_prefetch(&emask[node * Kn], 0, 1);  // epilogue data, fetched early
  __builtin_prefetch(&cnt[node], 0, 1);

  { // stage both weight layers up front (double buffer -> no mid-kernel staging)
    const uint4* s1 = (const uint4*)We1t;
    const uint4* s2 = (const uint4*)We2t;
    uint4* d1 = (uint4*)sW0;
    uint4* d2 = (uint4*)sW1;
    for (int i = tid; i < 128 * 16; i += 256) {
      int r = i >> 4, c = i & 15;             // 16 uint4 per 128-elem row
      d1[r * 17 + c] = s1[i];                 // PAD128 = 17 uint4 per row
      d2[r * 17 + c] = s2[i];
    }
  }
  { // build edge matrix: rows = node*16+nbr; cols 0..63 recv x, 64..127 gathered x
    int rowA = tid & 127, part = tid >> 7;
    int nd = nodeBase + (rowA >> 4);
    int src = (part == 0) ? nd : idx[nd * Kn + (rowA & 15)];
    const uint4* sp = (const uint4*)(xbf + (size_t)src * Fd);
    uint4* dp = (uint4*)(sA + rowA * PAD128 + part * 64);
    #pragma unroll
    for (int i = 0; i < 8; ++i) dp[i] = sp[i];
  }
  __syncthreads();

  v8f acc[8];
  { // GEMM1: wave w owns rows w*16..w*16+15; kc-outer, batched B loads
    v16bf af[4];
    #pragma unroll
    for (int kc = 0; kc < 4; ++kc) af[kc] = load_a_frag(sA, PAD128, wave * 16, kc * 32, lane);
    #pragma unroll
    for (int nt = 0; nt < 8; ++nt) acc[nt] = (v8f){};
    #pragma unroll
    for (int kc = 0; kc < 4; ++kc) {
      v16bf bf[8];
      #pragma unroll
      for (int nt = 0; nt < 8; ++nt) bf[nt] = load_b_frag(sW0, PAD128, nt * 16, kc * 32, lane);
      #pragma unroll
      for (int nt = 0; nt < 8; ++nt)
        acc[nt] = __builtin_amdgcn_wmma_f32_16x16x32_bf16(false, af[kc], false, bf[nt],
                                                          (short)0, acc[nt], false, false);
    }
  }
  // h1 = relu(acc + be1) -> own strip rows of sA
  #pragma unroll
  for (int nt = 0; nt < 8; ++nt) {
    int col = nt * 16 + (lane & 15);
    float bv = be1[col];
    #pragma unroll
    for (int v = 0; v < 8; ++v) {
      int rowA = wave * 16 + h * 8 + v;
      sA[rowA * PAD128 + col] = f2bf(fmaxf(acc[nt][v] + bv, 0.f));
    }
  }
  __syncthreads();                             // order h1 stores vs. b128 reloads
  { // GEMM2 (weights already staged in sW1)
    v16bf af[4];
    #pragma unroll
    for (int kc = 0; kc < 4; ++kc) af[kc] = load_a_frag(sA, PAD128, wave * 16, kc * 32, lane);
    #pragma unroll
    for (int nt = 0; nt < 8; ++nt) acc[nt] = (v8f){};
    #pragma unroll
    for (int kc = 0; kc < 4; ++kc) {
      v16bf bf[8];
      #pragma unroll
      for (int nt = 0; nt < 8; ++nt) bf[nt] = load_b_frag(sW1, PAD128, nt * 16, kc * 32, lane);
      #pragma unroll
      for (int nt = 0; nt < 8; ++nt)
        acc[nt] = __builtin_amdgcn_wmma_f32_16x16x32_bf16(false, af[kc], false, bf[nt],
                                                          (short)0, acc[nt], false, false);
    }
  }
  // epilogue: relu + bias, mask, mean-pool over the 16 neighbor rows (= one wave strip)
  float cv = cnt[node];
  float em[8];
  #pragma unroll
  for (int v = 0; v < 8; ++v) em[v] = emask[node * Kn + h * 8 + v];
  #pragma unroll
  for (int nt = 0; nt < 8; ++nt) {
    int col = nt * 16 + (lane & 15);
    float bv = be2[col];
    float s = 0.f;
    #pragma unroll
    for (int v = 0; v < 8; ++v) s += fmaxf(acc[nt][v] + bv, 0.f) * em[v];
    s += __shfl_xor(s, 16, 32);                // rows 0-7 + rows 8-15
    if (lane < 16) pooled[(size_t)node * Hd + col] = s / cv;
  }
}

// ---------------- node MLP: 128 nodes/block; (128x192)@(192x128) then (128x128)@(128x64) --
__global__ void node_mlp_kernel(const unsigned short* __restrict__ xbf,
                                const float* __restrict__ pooled,
                                const unsigned short* __restrict__ Wn1t,
                                const unsigned short* __restrict__ Wn2t,
                                const float* __restrict__ bn1, const float* __restrict__ bn2,
                                const int* __restrict__ mask, float* __restrict__ out) {
  __shared__ unsigned short sA[128 * PAD192];   // nh then h, padded
  __shared__ unsigned short sW1[Hd * PAD192];   // Wn1^T, padded
  __shared__ unsigned short sW2[FOd * PAD128];  // Wn2^T, padded
  int tid = threadIdx.x, wave = tid >> 5, lane = tid & 31;
  int h = lane >> 4;
  int nodeBase = blockIdx.x * 128;

  __builtin_prefetch(&mask[nodeBase + tid >= Bc * Nn ? 0 : nodeBase + tid], 0, 1);

  { // stage Wn1^T [128][192] and Wn2^T [64][128] up front
    const uint4* s1 = (const uint4*)Wn1t;
    uint4* d1 = (uint4*)sW1;
    for (int i = tid; i < 128 * 24; i += 256) {
      int r = i / 24, c = i % 24;             // 24 uint4 per 192-elem row
      d1[r * 25 + c] = s1[i];                 // PAD192 = 25 uint4 per row
    }
    const uint4* s2 = (const uint4*)Wn2t;
    uint4* d2 = (uint4*)sW2;
    for (int i = tid; i < 64 * 16; i += 256) {
      int r = i >> 4, c = i & 15;
      d2[r * 17 + c] = s2[i];
    }
  }
  { // build nh: cols 0..63 = x (bf16 copy), cols 64..191 = pooled (f32->bf16)
    int rowA = tid & 127, part = tid >> 7;
    int node = nodeBase + rowA;
    if (part == 0) {
      const uint4* sp = (const uint4*)(xbf + (size_t)node * Fd);
      uint4* dp = (uint4*)(sA + rowA * PAD192);
      #pragma unroll
      for (int i = 0; i < 8; ++i) dp[i] = sp[i];
      for (int c = 0; c < 32; ++c)
        sA[rowA * PAD192 + 64 + c] = f2bf(pooled[(size_t)node * Hd + c]);
    } else {
      for (int c = 32; c < 128; ++c)
        sA[rowA * PAD192 + 64 + c] = f2bf(pooled[(size_t)node * Hd + c]);
    }
  }
  __syncthreads();

  v8f acc[8];
  { // GEMM1: K=192 -> 6 chunks, kc-outer with batched B loads
    v16bf af[6];
    #pragma unroll
    for (int kc = 0; kc < 6; ++kc) af[kc] = load_a_frag(sA, PAD192, wave * 16, kc * 32, lane);
    #pragma unroll
    for (int nt = 0; nt < 8; ++nt) acc[nt] = (v8f){};
    #pragma unroll
    for (int kc = 0; kc < 6; ++kc) {
      v16bf bf[8];
      #pragma unroll
      for (int nt = 0; nt < 8; ++nt) bf[nt] = load_b_frag(sW1, PAD192, nt * 16, kc * 32, lane);
      #pragma unroll
      for (int nt = 0; nt < 8; ++nt)
        acc[nt] = __builtin_amdgcn_wmma_f32_16x16x32_bf16(false, af[kc], false, bf[nt],
                                                          (short)0, acc[nt], false, false);
    }
  }
  // h = relu(acc + bn1) -> own strip rows of sA (cols 0..127, stride stays PAD192)
  #pragma unroll
  for (int nt = 0; nt < 8; ++nt) {
    int col = nt * 16 + (lane & 15);
    float bv = bn1[col];
    #pragma unroll
    for (int v = 0; v < 8; ++v) {
      int rowA = wave * 16 + h * 8 + v;
      sA[rowA * PAD192 + col] = f2bf(fmaxf(acc[nt][v] + bv, 0.f));
    }
  }
  __syncthreads();                             // order h stores vs. b128 reloads
  v8f acc2[4];
  { // GEMM2: N=64 -> 4 tiles, K=128 -> 4 chunks
    v16bf af[4];
    #pragma unroll
    for (int kc = 0; kc < 4; ++kc) af[kc] = load_a_frag(sA, PAD192, wave * 16, kc * 32, lane);
    #pragma unroll
    for (int nt = 0; nt < 4; ++nt) acc2[nt] = (v8f){};
    #pragma unroll
    for (int kc = 0; kc < 4; ++kc) {
      v16bf bf[4];
      #pragma unroll
      for (int nt = 0; nt < 4; ++nt) bf[nt] = load_b_frag(sW2, PAD128, nt * 16, kc * 32, lane);
      #pragma unroll
      for (int nt = 0; nt < 4; ++nt)
        acc2[nt] = __builtin_amdgcn_wmma_f32_16x16x32_bf16(false, af[kc], false, bf[nt],
                                                           (short)0, acc2[nt], false, false);
    }
  }
  // out = (acc2 + bn2) * mask
  #pragma unroll
  for (int nt = 0; nt < 4; ++nt) {
    int col = nt * 16 + (lane & 15);
    float bv = bn2[col];
    #pragma unroll
    for (int v = 0; v < 8; ++v) {
      int node = nodeBase + wave * 16 + h * 8 + v;
      float mv = (float)mask[node];
      out[(size_t)node * FOd + col] = (acc2[nt][v] + bv) * mv;
    }
  }
}

extern "C" void kernel_launch(void* const* d_in, const int* in_sizes, int n_in,
                              void* d_out, int out_size, void* d_ws, size_t ws_size,
                              hipStream_t stream) {
  (void)in_sizes; (void)n_in; (void)out_size; (void)ws_size;
  const float* nodes = (const float*)d_in[0];
  const int*   mask  = (const int*)d_in[1];
  const float* We1   = (const float*)d_in[2];
  const float* be1   = (const float*)d_in[3];
  const float* We2   = (const float*)d_in[4];
  const float* be2   = (const float*)d_in[5];
  const float* Wn1   = (const float*)d_in[6];
  const float* bn1   = (const float*)d_in[7];
  const float* Wn2   = (const float*)d_in[8];
  const float* bn2   = (const float*)d_in[9];
  float* out = (float*)d_out;

  char* ws = (char*)d_ws;
  size_t off = 0;
  auto alloc = [&](size_t bytes) -> void* {
    void* p = ws + off;
    off += (bytes + 255) & ~(size_t)255;
    return p;
  };
  const int rows = Bc * Nn;
  float*          x      = (float*)alloc((size_t)rows * Fd * 4);
  float*          sq     = (float*)alloc((size_t)rows * 4);
  unsigned short* xbf    = (unsigned short*)alloc((size_t)rows * Fd * 2);
  int*            idx    = (int*)alloc((size_t)rows * Kn * 4);
  float*          emaskA = (float*)alloc((size_t)rows * Kn * 4);
  float*          cntA   = (float*)alloc((size_t)rows * 4);
  float*          pooled = (float*)alloc((size_t)rows * Hd * 4);
  unsigned short* We1t   = (unsigned short*)alloc((size_t)(2 * Fd) * Hd * 2);
  unsigned short* We2t   = (unsigned short*)alloc((size_t)Hd * Hd * 2);
  unsigned short* Wn1t   = (unsigned short*)alloc((size_t)(Fd + Hd) * Hd * 2);
  unsigned short* Wn2t   = (unsigned short*)alloc((size_t)Hd * FOd * 2);

  prep_x_kernel<<<rows / 8, 256, 0, stream>>>(nodes, mask, x, xbf, sq);
  wprep_kernel<<<(2 * Fd * Hd + 255) / 256, 256, 0, stream>>>(We1, We1t, 2 * Fd, Hd);
  wprep_kernel<<<(Hd * Hd + 255) / 256, 256, 0, stream>>>(We2, We2t, Hd, Hd);
  wprep_kernel<<<((Fd + Hd) * Hd + 255) / 256, 256, 0, stream>>>(Wn1, Wn1t, Fd + Hd, Hd);
  wprep_kernel<<<(Hd * FOd + 255) / 256, 256, 0, stream>>>(Wn2, Wn2t, Hd, FOd);
  topk_kernel<<<rows / 8, 256, 0, stream>>>(x, sq, mask, idx, emaskA, cntA);
  edge_mlp_kernel<<<rows / 8, 256, 0, stream>>>(xbf, idx, We1t, We2t, be1, be2,
                                                emaskA, cntA, pooled);
  node_mlp_kernel<<<rows / 128, 256, 0, stream>>>(xbf, pooled, Wn1t, Wn2t, bn1, bn2,
                                                  mask, out);
}
